// FusedQKVFP8Linear_70403103916686
// MI455X (gfx1250) — 8-bit matrix operand order repaired and checked by reading, compiled, not run on hardware
//
#include <hip/hip_runtime.h>

#define MDIM 8192
#define KDIM 3072
#define NDIM 9216
#define SEGN 3072
#define LSTRA 136
#define LSTRB 144

typedef __attribute__((ext_vector_type(16))) int          v16i;
typedef __attribute__((ext_vector_type(8)))  float        v8f;
typedef __attribute__((ext_vector_type(8)))  int          v8i;
typedef __attribute__((ext_vector_type(4)))  int          v4i;
typedef __attribute__((ext_vector_type(4)))  unsigned int v4u;

#if defined(__AMDGCN__) && __has_builtin(__builtin_amdgcn_tensor_load_to_lds) && \
    __has_builtin(__builtin_amdgcn_s_wait_tensorcnt)
#define USE_TDM 1
#else
#define USE_TDM 0
#endif

__device__ __forceinline__ unsigned short f2bf(float f) {
  unsigned u = __float_as_uint(f);
  u += 0x7fffu + ((u >> 16) & 1u);
  return (unsigned short)(u >> 16);
}

__device__ __forceinline__ unsigned char f32_to_e4m3(float f) {
  unsigned u = __float_as_uint(f);
  unsigned s = (u >> 24) & 0x80u;
  unsigned a = u & 0x7FFFFFFFu;
  if (a > 0x43E00000u) a = 0x43E00000u;
  if (a < 0x3C800000u) {
    float q = __uint_as_float(a) * 512.0f;
    int qi = (int)rintf(q);
    return (unsigned char)(s | (unsigned)(qi & 0x7F));
  }
  unsigned r = a + 0x0007FFFFu + ((a >> 20) & 1u);
  int e = (int)(r >> 23) - 127 + 7;
  unsigned m = (r >> 20) & 7u;
  if (e > 15) { e = 15; m = 6u; }
  return (unsigned char)(s | ((unsigned)e << 3) | m);
}

__global__ __launch_bounds__(256) void k_transpose_w(const unsigned char* __restrict__ w,
                                                     unsigned char* __restrict__ wt) {
  __shared__ unsigned char t[64 * 80];
  const int n0 = blockIdx.x * 64;
  const int k0 = blockIdx.y * 64;
  const int tid = threadIdx.x;
  {
    int r = tid >> 2;
    int c = (tid & 3) << 4;
    *(int4*)(t + r * 80 + c) =
        *(const int4*)(w + (size_t)(k0 + r) * NDIM + n0 + c);
  }
  __syncthreads();
  {
    int n = tid >> 2;
    int kc = (tid & 3) << 4;
    alignas(16) unsigned char buf[16];
#pragma unroll
    for (int j = 0; j < 16; ++j) buf[j] = t[(kc + j) * 80 + n];
    *(int4*)(wt + (size_t)(n0 + n) * KDIM + k0 + kc) = *(int4*)buf;
  }
}

__global__ __launch_bounds__(256) void k_quant_x(const float* __restrict__ x,
                                                 unsigned char* __restrict__ xq,
                                                 float* __restrict__ scales) {
  const int row = blockIdx.x;
  const int tid = threadIdx.x;
  const float4* xr = (const float4*)(x + (size_t)row * KDIM);
  float4 v[3];
  float am = 0.0f;
#pragma unroll
  for (int i = 0; i < 3; ++i) {
    v[i] = xr[tid + i * 256];
    am = fmaxf(am, fmaxf(fmaxf(fabsf(v[i].x), fabsf(v[i].y)),
                         fmaxf(fabsf(v[i].z), fabsf(v[i].w))));
  }
#pragma unroll
  for (int off = 16; off > 0; off >>= 1)
    am = fmaxf(am, __shfl_xor(am, off, 32));
  __shared__ float sm[8];
  if ((tid & 31) == 0) sm[tid >> 5] = am;
  __syncthreads();
  float m = sm[0];
#pragma unroll
  for (int i = 1; i < 8; ++i) m = fmaxf(m, sm[i]);
  m = fmaxf(m, 1e-12f);
  const float a_scale = m * (1.0f / 448.0f);
  const float inv = 448.0f / m;
  if (tid == 0) scales[row] = a_scale;

  unsigned* oq = (unsigned*)(xq + (size_t)row * KDIM);
#pragma unroll
  for (int i = 0; i < 3; ++i) {
    float a0 = v[i].x * inv, a1 = v[i].y * inv;
    float a2 = v[i].z * inv, a3 = v[i].w * inv;
#if defined(__AMDGCN__) && __has_builtin(__builtin_amdgcn_cvt_pk_fp8_f32)
    int p = 0;
    p = __builtin_amdgcn_cvt_pk_fp8_f32(a0, a1, p, false);
    p = __builtin_amdgcn_cvt_pk_fp8_f32(a2, a3, p, true);
    oq[tid + i * 256] = (unsigned)p;
#else
    unsigned p = (unsigned)f32_to_e4m3(a0) | ((unsigned)f32_to_e4m3(a1) << 8) |
                 ((unsigned)f32_to_e4m3(a2) << 16) |
                 ((unsigned)f32_to_e4m3(a3) << 24);
    oq[tid + i * 256] = p;
#endif
  }
}

__device__ __forceinline__ void compute_tile(const unsigned char* aRow0,
                                             const unsigned char* aRow1,
                                             const unsigned char* bCol,
                                             v8f acc[8]) {
  v16i a0, a1;
#pragma unroll
  for (int p = 0; p < 8; ++p) {
    unsigned long long q0 = *(const unsigned long long*)(aRow0 + 32 * (p >> 1) + 8 * (p & 1));
    unsigned long long q1 = *(const unsigned long long*)(aRow1 + 32 * (p >> 1) + 8 * (p & 1));
    a0[2 * p] = (int)(unsigned)q0;
    a0[2 * p + 1] = (int)(q0 >> 32);
    a1[2 * p] = (int)(unsigned)q1;
    a1[2 * p + 1] = (int)(q1 >> 32);
  }
#pragma unroll
  for (int j = 0; j < 4; ++j) {
    v16i b;
    const unsigned char* bp = bCol + j * 16 * LSTRB;
#pragma unroll
    for (int g = 0; g < 4; ++g) {
      int4 q = *(const int4*)(bp + 32 * g);
      b[4 * g] = q.x;
      b[4 * g + 1] = q.y;
      b[4 * g + 2] = q.z;
      b[4 * g + 3] = q.w;
    }
    acc[2 * j] = __builtin_amdgcn_wmma_f32_16x16x128_fp8_fp8(
        a0, b, (short)0, acc[2 * j], false, false);
    acc[2 * j + 1] = __builtin_amdgcn_wmma_f32_16x16x128_fp8_fp8(
        a1, b, (short)0, acc[2 * j + 1], false, false);
  }
}

#if USE_TDM
__device__ __forceinline__ void tdm_load_tile(unsigned lds_addr,
                                              unsigned long long gaddr,
                                              unsigned g1w0, unsigned rows) {
  v4u g0;
  g0.x = 1u;
  g0.y = lds_addr;
  g0.z = (unsigned)gaddr;
  g0.w = (unsigned)((gaddr >> 32) & 0x01FFFFFFull) | (2u << 30);
  v8i g1;
  g1[0] = (int)g1w0;
  g1[1] = (int)((unsigned)KDIM << 16);
  g1[2] = (int)(rows << 16);
  g1[3] = (int)(128u << 16);
  g1[4] = 128;
  g1[5] = KDIM;
  g1[6] = 0;
  g1[7] = 0;
  v4i z4 = {0, 0, 0, 0};
  v8i z8 = {0, 0, 0, 0, 0, 0, 0, 0};
  __builtin_amdgcn_tensor_load_to_lds(g0, g1, z4, z4, z8, 0);
}
#endif

__global__ __launch_bounds__(256) void k_gemm(const unsigned char* __restrict__ xq,
                                              const unsigned char* __restrict__ wt,
                                              const float* __restrict__ scales,
                                              const float* __restrict__ w_scale,
                                              const float* __restrict__ bias,
                                              unsigned short* __restrict__ out) {
  __shared__ unsigned char sA[2][128 * LSTRA];
  __shared__ unsigned char sB[2][128 * LSTRB];

  const int tid = threadIdx.x;
  const int lane = tid & 31;
  const int wave = tid >> 5;
  const int wm = wave & 3;
  const int wn = wave >> 2;
  const int hf = lane >> 4;
  const int l15 = lane & 15;
  const int m0 = blockIdx.y * 128;
  const int n0 = blockIdx.x * 128;

  v8f acc[8];
  const v8f vzero = {0.f, 0.f, 0.f, 0.f, 0.f, 0.f, 0.f, 0.f};
#pragma unroll
  for (int i = 0; i < 8; ++i) acc[i] = vzero;

  const unsigned aOff = (unsigned)((wm * 32 + l15) * LSTRA + 16 * hf);
  const unsigned bOff = (unsigned)((wn * 64 + l15) * LSTRB + 16 * hf);

  const int nT = KDIM / 128;

#if USE_TDM
  const bool w0 = (wave == 0);
  const unsigned ldsA0 = (unsigned)(uintptr_t)&sA[0][0];
  const unsigned ldsA1 = (unsigned)(uintptr_t)&sA[1][0];
  const unsigned ldsB0 = (unsigned)(uintptr_t)&sB[0][0];
  const unsigned ldsB1 = (unsigned)(uintptr_t)&sB[1][0];
  const unsigned g1A = (1u << 20) | (4u << 22) | (1u << 25);
  const unsigned g1B = (1u << 20) | (4u << 22) | (3u << 25);
  const unsigned long long gA =
      (unsigned long long)(uintptr_t)xq + (unsigned long long)m0 * KDIM;
  const unsigned long long gB =
      (unsigned long long)(uintptr_t)wt + (unsigned long long)n0 * KDIM;

  if (w0) {
    tdm_load_tile(ldsA0, gA, g1A, MDIM);
    tdm_load_tile(ldsB0, gB, g1B, NDIM);
  }
  for (int it = 0; it < nT; ++it) {
    const int cur = it & 1;
    if (w0) {
      if (it + 1 < nT) {
        const unsigned long long ko = (unsigned long long)(it + 1) * 128ull;
        tdm_load_tile(cur ? ldsA0 : ldsA1, gA + ko, g1A, MDIM);
        tdm_load_tile(cur ? ldsB0 : ldsB1, gB + ko, g1B, NDIM);
        __builtin_amdgcn_s_wait_tensorcnt(2);
      } else {
        __builtin_amdgcn_s_wait_tensorcnt(0);
      }
    }
    __syncthreads();
    compute_tile(&sA[cur][0] + aOff, &sA[cur][0] + aOff + 16 * LSTRA,
                 &sB[cur][0] + bOff, acc);
    __syncthreads();
  }
#else
  for (int it = 0; it < nT; ++it) {
    const int k0 = it * 128;
#pragma unroll
    for (int i = 0; i < 4; ++i) {
      int id = tid + i * 256;
      int r = id >> 3;
      int c = (id & 7) << 4;
      *(int4*)(&sA[0][0] + r * LSTRA + c) =
          *(const int4*)(xq + (size_t)(m0 + r) * KDIM + k0 + c);
      *(int4*)(&sB[0][0] + r * LSTRB + c) =
          *(const int4*)(wt + (size_t)(n0 + r) * KDIM + k0 + c);
    }
    __syncthreads();
    compute_tile(&sA[0][0] + aOff, &sA[0][0] + aOff + 16 * LSTRA,
                 &sB[0][0] + bOff, acc);
    __syncthreads();
  }
#endif

  float sr0[8], sr1[8];
#pragma unroll
  for (int r = 0; r < 8; ++r) {
    sr0[r] = scales[m0 + wm * 32 + r + 8 * hf];
    sr1[r] = scales[m0 + wm * 32 + 16 + r + 8 * hf];
  }
#pragma unroll
  for (int j = 0; j < 4; ++j) {
    int ncol = n0 + wn * 64 + j * 16 + l15;
    float wsc = w_scale[ncol];
    float bi = bias[ncol];
    int seg = ncol / SEGN;
    int nloc = ncol - seg * SEGN;
    unsigned short* op = out + (size_t)seg * MDIM * SEGN + nloc;
#pragma unroll
    for (int r = 0; r < 8; ++r) {
      int mr = m0 + wm * 32 + r + 8 * hf;
      op[(size_t)mr * SEGN] = f2bf(acc[2 * j][r] * (sr0[r] * wsc) + bi);
      op[(size_t)(mr + 16) * SEGN] =
          f2bf(acc[2 * j + 1][r] * (sr1[r] * wsc) + bi);
    }
  }
}

extern "C" void kernel_launch(void* const* d_in, const int* in_sizes, int n_in,
                              void* d_out, int out_size, void* d_ws,
                              size_t ws_size, hipStream_t stream) {
  (void)in_sizes; (void)n_in; (void)out_size; (void)ws_size;
  const float* x = (const float*)d_in[0];
  const unsigned char* w8 = (const unsigned char*)d_in[1];
  const float* w_scale = (const float*)d_in[2];
  const float* bias = (const float*)d_in[3];
  unsigned short* out = (unsigned short*)d_out;

  unsigned char* ws = (unsigned char*)d_ws;
  unsigned char* xq = ws;
  float* scales = (float*)(ws + (size_t)MDIM * KDIM);
  unsigned char* wt = ws + (size_t)MDIM * KDIM + 32768;

  k_transpose_w<<<dim3(NDIM / 64, KDIM / 64), 256, 0, stream>>>(w8, wt);
  k_quant_x<<<dim3(MDIM), 256, 0, stream>>>(x, xq, scales);
  k_gemm<<<dim3(NDIM / 128, MDIM / 128), 256, 0, stream>>>(xq, wt, scales,
                                                           w_scale, bias, out);
}
